// PmeGNN_41523743817954
// MI455X (gfx1250) — matrix operand order repaired and checked by reading, not compiled, not run on hardware
//
#include <hip/hip_runtime.h>
#include <hip/hip_bf16.h>
#include <cstdint>
#include <cstddef>

typedef _Float16 f16;
typedef __attribute__((ext_vector_type(16))) _Float16 v16h;
typedef __attribute__((ext_vector_type(8)))  _Float16 v8h;
typedef __attribute__((ext_vector_type(8)))  float    v8f;

#define DEV __device__ __forceinline__

DEV float sigm_(float x){ return 1.f/(1.f+__expf(-x)); }
DEV float silu_(float x){ return x*sigm_(x); }
DEV float gelu_(float x){ return 0.5f*x*(1.f+tanhf(0.7978845608f*(x+0.044715f*x*x*x))); }

constexpr int NATOM = 2048;
constexpr int NE    = 32768;
constexpr int NE2   = 65536;
constexpr int NB    = 8;
constexpr int NP    = 20;
constexpr int NMESH = 64000;          // NB*NP^3
constexpr int NFEAT = 144;
constexpr int NRE   = 64;
constexpr int NRH   = 128;
constexpr int NMC   = 32;
constexpr int NMH   = 16;
constexpr int NKZ   = 10;
constexpr int CPLXN = NB*NP*NP*NKZ*NMH; // 512000 complex points

// ---------------------------------------------------------------------------
// WMMA GEMM: C(MxN) = act( A(M x Kp, f16, lda=Kp) @ B(Kp x N, f16) )
// Kp is a multiple of 32; B pad rows are zeroed so A pad columns are inert.
// One wave -> one 16x16 tile; VGPR layouts per CDNA5 ISA 7.12.2.
// A loads are unconditional 16-byte vectors (global_load_b128); B is read per 2-byte element.
// ---------------------------------------------------------------------------
union AU { v16h v; v8h q[2]; f16 h[16]; };
union CU { v8f  v; float f[8]; };

template<int ACT, int OUTH>
__global__ __launch_bounds__(256)
void gemm16(const f16* __restrict__ A, const f16* __restrict__ Bw,
            void* __restrict__ Cout, int Mr, int Nc, int Kp)
{
  const int wid  = threadIdx.x >> 5;
  const int lane = threadIdx.x & 31;
  const int tilesN = Nc >> 4;
  const int tilesM = Mr >> 4;
  const int tile = blockIdx.x * (blockDim.x >> 5) + wid;
  if (tile >= tilesM * tilesN) return;
  const int tm = tile / tilesN, tn = tile % tilesN;

  const int m  = tm*16 + (lane & 15);          // A: both lane halves hold M=lane%16
  const int kb = (lane >> 4) * 8;              // K sub-base per lane-half
  CU c;
  #pragma unroll
  for (int i = 0; i < 8; ++i) c.f[i] = 0.f;

  const f16* arow  = A  + (size_t)m * Kp + kb;
  const f16* bbase = Bw + (size_t)kb * Nc + tn*16 + (lane & 15);   // B: lane%16 = N col
  for (int k0 = 0; k0 < Kp; k0 += 32) {
    AU a, b;
    a.q[0] = *(const v8h*)(arow + k0);          // K = k0+kb   .. +7
    a.q[1] = *(const v8h*)(arow + k0 + 16);     // K = k0+kb+16.. +23
    const f16* brow = bbase + (size_t)k0 * Nc;
    #pragma unroll
    for (int s = 0; s < 8; ++s) {               // K = k0+kb+s, k0+kb+16+s
      b.h[s]     = brow[(size_t)s * Nc];
      b.h[8 + s] = brow[(size_t)(16 + s) * Nc];
    }
    c.v = __builtin_amdgcn_wmma_f32_16x16x32_f16(false, a.v, false, b.v,
                                                 (short)0, c.v, false, false);
  }

  const int row0 = tm*16 + ((lane >= 16) ? 8 : 0);  // C/D: VGPR r -> M=r / r+8
  const int col  = tn*16 + (lane & 15);
  #pragma unroll
  for (int r = 0; r < 8; ++r) {
    float v = c.f[r];
    if (ACT == 1) v = silu_(v);
    else if (ACT == 2) v = gelu_(v);
    const size_t o = (size_t)(row0 + r) * Nc + col;
    if (OUTH) ((f16*)Cout)[o] = (f16)v;
    else      ((float*)Cout)[o] = v;
  }
}

// ---------------------------------------------------------------------------
// staging / element-wise kernels
// ---------------------------------------------------------------------------
__global__ void k_f32_to_f16(const float* __restrict__ s, f16* __restrict__ d, int n){
  int i = blockIdx.x*blockDim.x + threadIdx.x; if (i < n) d[i] = (f16)s[i];
}
// A staging with K padded to Kp (zero pad)
__global__ void k_padA(const float* __restrict__ s, f16* __restrict__ d,
                       int K, int Kp, int n){
  int i = blockIdx.x*blockDim.x + threadIdx.x; if (i >= n) return;
  int m = i / Kp, k = i % Kp;
  d[i] = (k < K) ? (f16)s[(size_t)m*K + k] : (f16)0.f;
}
// weight staging: Kp x Nc, rows >= K zeroed (flat layout matches src for r < K)
__global__ void k_padW(const float* __restrict__ s, f16* __restrict__ d,
                       int K, int Nc, int n){
  int i = blockIdx.x*blockDim.x + threadIdx.x; if (i >= n) return;
  int r = i / Nc;
  d[i] = (r < K) ? (f16)s[i] : (f16)0.f;
}
__global__ void k_zero(float* p, int n){
  int i = blockIdx.x*blockDim.x + threadIdx.x; if (i < n) p[i] = 0.f;
}
__global__ void k_sigabs(float* p, int n){
  int i = blockIdx.x*blockDim.x + threadIdx.x; if (i < n){ float v = p[i]; p[i] = v * sigm_(fabsf(v)); }
}
__global__ void k_gelu_sum(const float* a, const float* b, float* o, int n){
  int i = blockIdx.x*blockDim.x + threadIdx.x; if (i < n) o[i] = gelu_(a[i] + b[i]);
}
__global__ void k_addto(float* a, const float* b, int n){
  int i = blockIdx.x*blockDim.x + threadIdx.x; if (i < n) a[i] += b[i];
}

DEV void sh9(float x, float y, float z, float* s){
  s[0] = 0.28209479177387814f;
  s[1] = 0.4886025119029199f * y;
  s[2] = 0.4886025119029199f * z;
  s[3] = 0.4886025119029199f * x;
  s[4] = 1.0925484305920792f * x * y;
  s[5] = 1.0925484305920792f * y * z;
  s[6] = 0.31539156525252005f * (3.f*z*z - 1.f);
  s[7] = 1.0925484305920792f * x * z;
  s[8] = 0.5462742152960396f * (x*x - y*y);
}
DEV void gauss64(float r, f16* out){
  const float step = 5.f / 63.f;
  #pragma unroll 8
  for (int j = 0; j < NRE; ++j){
    float d = (r - j*step) / step;
    out[j] = (f16)(__expf(-d*d) * (8.f/1.12f));
  }
}

__global__ void k_edge1(const float* __restrict__ pos, const int* __restrict__ src,
                        const int* __restrict__ dst, float* __restrict__ sh,
                        f16* __restrict__ rad, int n){
  int e = blockIdx.x*blockDim.x + threadIdx.x; if (e >= n) return;
  int a = src[e], b = dst[e];
  float ex = pos[a*3+0]-pos[b*3+0], ey = pos[a*3+1]-pos[b*3+1], ez = pos[a*3+2]-pos[b*3+2];
  float el = sqrtf(ex*ex+ey*ey+ez*ez) + 1e-8f;
  float inv = 1.f/el;
  float s[9]; sh9(ex*inv, ey*inv, ez*inv, s);
  #pragma unroll
  for (int q = 0; q < 9; ++q) sh[(size_t)e*9 + q] = s[q];
  gauss64(el, rad + (size_t)e*NRE);
}

__global__ void k_edge2(const float* __restrict__ pos, const float* __restrict__ cell,
                        const int* __restrict__ asrc, const int* __restrict__ mdst,
                        float* __restrict__ sh, f16* __restrict__ rad, int n){
  int e = blockIdx.x*blockDim.x + threadIdx.x; if (e >= n) return;
  int m = mdst[e];
  int b = m / 8000, rem = m % 8000;
  float fg0 = (float)(rem/400) / 20.f;
  float fg1 = (float)((rem/20)%20) / 20.f;
  float fg2 = (float)(rem%20) / 20.f;
  const float* C = cell + b*9;
  float mp0 = fg0*C[0] + fg1*C[3] + fg2*C[6];
  float mp1 = fg0*C[1] + fg1*C[4] + fg2*C[7];
  float mp2 = fg0*C[2] + fg1*C[5] + fg2*C[8];
  int a = asrc[e];
  float ex = mp0 - pos[a*3+0], ey = mp1 - pos[a*3+1], ez = mp2 - pos[a*3+2];
  float el = sqrtf(ex*ex+ey*ey+ez*ez) + 1e-8f;
  float inv = 1.f/el;
  float s[9]; sh9(ex*inv, ey*inv, ez*inv, s);
  #pragma unroll
  for (int q = 0; q < 9; ++q) sh[(size_t)e*9 + q] = s[q];
  gauss64(el, rad + (size_t)e*NRE);
}

// gather emb rows (16-wide) into Kp=32-padded f16 staging
__global__ void k_gather_emb(const float* __restrict__ emb, const int* __restrict__ z,
                             f16* __restrict__ xh, int n){
  int i = blockIdx.x*blockDim.x + threadIdx.x; if (i >= n) return;
  int node = i >> 5, f = i & 31;
  xh[i] = (f < 16) ? (f16)emb[z[node]*NMH + f] : (f16)0.f;
}

__global__ void k_norm_act(const float* __restrict__ xin, float* __restrict__ xout, int n){
  int i = blockIdx.x*blockDim.x + threadIdx.x; if (i >= n) return;
  int node = i >> 4, r = i & 15;
  const float* xi = xin + (size_t)node*NFEAT;
  float* xo = xout + (size_t)node*NFEAT;
  float s = xi[r];
  xo[r] = s * sigm_(fabsf(s));
  float v0 = xi[16 + r*3+0], v1 = xi[16 + r*3+1], v2 = xi[16 + r*3+2];
  float g = sigm_(sqrtf(v0*v0 + v1*v1 + v2*v2));
  xo[16 + r*3+0] = v0*g; xo[16 + r*3+1] = v1*g; xo[16 + r*3+2] = v2*g;
  float t[5], tn = 0.f;
  #pragma unroll
  for (int j = 0; j < 5; ++j){ t[j] = xi[64 + r*5 + j]; tn += t[j]*t[j]; }
  float gt = sigm_(sqrtf(tn));
  #pragma unroll
  for (int j = 0; j < 5; ++j) xo[64 + r*5 + j] = t[j]*gt;
}

// msg[e,o] = (sum_s sh[e,s] * Y[src[e], s*O+o]) * radout[e,o]; atomic scatter to out[dst[e],o]
template<int O>
__global__ void k_msg_scatter(const float* __restrict__ Y, const float* __restrict__ sh,
                              const float* __restrict__ radout, const int* __restrict__ src,
                              const int* __restrict__ dst, float* __restrict__ out, int nE){
  int idx = blockIdx.x*blockDim.x + threadIdx.x;
  if (idx >= nE * O) return;
  int e = idx / O, o = idx % O;
  const float* y = Y + (size_t)src[e] * (9*O) + o;
  const float* s = sh + (size_t)e*9;
  float acc = 0.f;
  #pragma unroll
  for (int q = 0; q < 9; ++q) acc += s[q] * y[q*O];
  acc *= radout[(size_t)e*O + o];
  atomicAdd(out + (size_t)dst[e]*O + o, acc);
}

__global__ void k_fc0(const float* __restrict__ mf, const float* __restrict__ W,
                      const float* __restrict__ bv, float* __restrict__ h, int n){
  int idx = blockIdx.x*blockDim.x + threadIdx.x; if (idx >= n) return;
  int m = idx >> 4, o = idx & 15;
  int rem = m % 8000;
  float gx = (float)(rem/400)/20.f, gy = (float)((rem/20)%20)/20.f, gz = (float)(rem%20)/20.f;
  float acc = bv[o];
  const float* mrow = mf + (size_t)m*NMC;
  #pragma unroll 8
  for (int f = 0; f < NMC; ++f) acc += mrow[f] * W[f*NMH + o];
  acc += gx*W[32*NMH+o] + gy*W[33*NMH+o] + gz*W[34*NMH+o];
  h[idx] = acc;
}

// ---------------------------------------------------------------------------
// separable 20-point DFT passes (only kz=0..9 survive the spectral weights)
// ---------------------------------------------------------------------------
__global__ void k_fft_z(const float* __restrict__ h, float2* __restrict__ o, int n){
  int idx = blockIdx.x*blockDim.x + threadIdx.x; if (idx >= n) return;
  int c = idx & 15; int t = idx >> 4;
  int kz = t % NKZ; t /= NKZ; int y = t % NP; t /= NP; int x = t % NP; int b = t / NP;
  const float* base = h + ((size_t)(((b*NP+x)*NP+y)*NP))*NMH + c;
  float re = 0.f, im = 0.f;
  for (int z = 0; z < NP; ++z){
    float ang = -0.31415926535f * (float)(z*kz);
    float s, ct; __sincosf(ang, &s, &ct);
    float v = base[z*NMH];
    re += v*ct; im += v*s;
  }
  float2 r; r.x = re; r.y = im; o[idx] = r;
}
__global__ void k_fft_y(const float2* __restrict__ in, float2* __restrict__ o, int n){
  int idx = blockIdx.x*blockDim.x + threadIdx.x; if (idx >= n) return;
  int c = idx & 15; int t = idx >> 4;
  int kz = t % NKZ; t /= NKZ; int ky = t % NP; t /= NP; int x = t % NP; int b = t / NP;
  float re = 0.f, im = 0.f;
  for (int y = 0; y < NP; ++y){
    float ang = -0.31415926535f * (float)(y*ky);
    float s, ct; __sincosf(ang, &s, &ct);
    float2 v = in[((size_t)(((b*NP+x)*NP+y)*NKZ+kz))*NMH + c];
    re += v.x*ct - v.y*s; im += v.x*s + v.y*ct;
  }
  float2 r; r.x = re; r.y = im; o[idx] = r;
}
__global__ void k_fft_x(const float2* __restrict__ in, float2* __restrict__ o, int n){
  int idx = blockIdx.x*blockDim.x + threadIdx.x; if (idx >= n) return;
  int c = idx & 15; int t = idx >> 4;
  int kz = t % NKZ; t /= NKZ; int ky = t % NP; t /= NP; int kx = t % NP; int b = t / NP;
  float re = 0.f, im = 0.f;
  for (int x = 0; x < NP; ++x){
    float ang = -0.31415926535f * (float)(x*kx);
    float s, ct; __sincosf(ang, &s, &ct);
    float2 v = in[((size_t)(((b*NP+x)*NP+ky)*NKZ+kz))*NMH + c];
    re += v.x*ct - v.y*s; im += v.x*s + v.y*ct;
  }
  float2 r; r.x = re; r.y = im; o[idx] = r;
}
__global__ void k_spec(const float2* __restrict__ X, float2* __restrict__ O,
                       const float* __restrict__ wr, const float* __restrict__ wi, int n){
  int idx = blockIdx.x*blockDim.x + threadIdx.x; if (idx >= n) return;
  int o = idx & 15; int t = idx >> 4;
  int kz = t % NKZ; t /= NKZ; int ky = t % NP; t /= NP; int kx = t % NP; int b = t / NP;
  int corner = (kx >= 10 ? 1 : 0) + (ky >= 10 ? 2 : 0);
  int wx = kx % 10, wy = ky % 10;
  const float2* xin = X + ((size_t)(((b*NP+kx)*NP+ky)*NKZ+kz))*NMH;
  float re = 0.f, im = 0.f;
  #pragma unroll 4
  for (int i = 0; i < 16; ++i){
    size_t widx = ((size_t)((corner*16+i)*16 + o))*1000 + wx*100 + wy*10 + kz;
    float a = wr[widx], bb = wi[widx];
    float2 xv = xin[i];
    re += a*xv.x - bb*xv.y;
    im += a*xv.y + bb*xv.x;
  }
  float2 r; r.x = re; r.y = im; O[idx] = r;
}
__global__ void k_ifft_x(const float2* __restrict__ in, float2* __restrict__ o, int n){
  int idx = blockIdx.x*blockDim.x + threadIdx.x; if (idx >= n) return;
  int c = idx & 15; int t = idx >> 4;
  int kz = t % NKZ; t /= NKZ; int ky = t % NP; t /= NP; int x = t % NP; int b = t / NP;
  float re = 0.f, im = 0.f;
  for (int kx = 0; kx < NP; ++kx){
    float ang = 0.31415926535f * (float)(kx*x);
    float s, ct; __sincosf(ang, &s, &ct);
    float2 v = in[((size_t)(((b*NP+kx)*NP+ky)*NKZ+kz))*NMH + c];
    re += v.x*ct - v.y*s; im += v.x*s + v.y*ct;
  }
  float2 r; r.x = re*0.05f; r.y = im*0.05f; o[idx] = r;
}
__global__ void k_ifft_y(const float2* __restrict__ in, float2* __restrict__ o, int n){
  int idx = blockIdx.x*blockDim.x + threadIdx.x; if (idx >= n) return;
  int c = idx & 15; int t = idx >> 4;
  int kz = t % NKZ; t /= NKZ; int y = t % NP; t /= NP; int x = t % NP; int b = t / NP;
  float re = 0.f, im = 0.f;
  for (int ky = 0; ky < NP; ++ky){
    float ang = 0.31415926535f * (float)(ky*y);
    float s, ct; __sincosf(ang, &s, &ct);
    float2 v = in[((size_t)(((b*NP+x)*NP+ky)*NKZ+kz))*NMH + c];
    re += v.x*ct - v.y*s; im += v.x*s + v.y*ct;
  }
  float2 r; r.x = re*0.05f; r.y = im*0.05f; o[idx] = r;
}
__global__ void k_ifft_z(const float2* __restrict__ in, float* __restrict__ outr, int n){
  int idx = blockIdx.x*blockDim.x + threadIdx.x; if (idx >= n) return;
  int c = idx & 15; int t = idx >> 4;
  int z = t % NP; t /= NP; int y = t % NP; t /= NP; int x = t % NP; int b = t / NP;
  const float2* base = in + ((size_t)(((b*NP+x)*NP+y)*NKZ))*NMH + c;
  float acc = base[0].x;                 // kz=0 (irfft drops Im)
  for (int kz = 1; kz < NKZ; ++kz){
    float ang = 0.31415926535f * (float)(kz*z);
    float s, ct; __sincosf(ang, &s, &ct);
    float2 v = base[(size_t)kz*NMH];
    acc += 2.f*(v.x*ct - v.y*s);
  }
  outr[idx] = acc * 0.05f;
}

// ---------------------------------------------------------------------------
// mesh->atom readout (O=1 paths, VALU)
// ---------------------------------------------------------------------------
__global__ void k_Ym(const float* __restrict__ h, const float* __restrict__ W,
                     float* __restrict__ Ym, int n){
  int idx = blockIdx.x*blockDim.x + threadIdx.x; if (idx >= n) return;
  int m = idx / 9, s = idx % 9;
  float acc = 0.f;
  const float* hr = h + (size_t)m*NMH;
  #pragma unroll
  for (int f = 0; f < 16; ++f) acc += hr[f] * W[f*9 + s];
  Ym[idx] = acc;
}
__global__ void k_dot_m3(const f16* __restrict__ h2, const float* __restrict__ w,
                         float* __restrict__ out, int n){
  int e = blockIdx.x*blockDim.x + threadIdx.x; if (e >= n) return;
  const f16* r = h2 + (size_t)e*NRH;
  float acc = 0.f;
  #pragma unroll 8
  for (int k = 0; k < NRH; ++k) acc += (float)r[k] * w[k];
  out[e] = acc;
}
__global__ void k_m2a_scatter(const float* __restrict__ Ym, const float* __restrict__ sh,
                              const float* __restrict__ rad, const int* __restrict__ mdst,
                              const int* __restrict__ asrc, float* __restrict__ af, int n){
  int e = blockIdx.x*blockDim.x + threadIdx.x; if (e >= n) return;
  const float* y = Ym + (size_t)mdst[e]*9;
  const float* s = sh + (size_t)e*9;
  float acc = 0.f;
  #pragma unroll
  for (int q = 0; q < 9; ++q) acc += s[q]*y[q];
  atomicAdd(af + asrc[e], acc * rad[e]);
}
__global__ void k_batch_reduce(const float* __restrict__ af, const int* __restrict__ batch,
                               float* __restrict__ out, int n){
  int i = blockIdx.x*blockDim.x + threadIdx.x; if (i >= n) return;
  atomicAdd(out + batch[i], af[i]);
}

// ---------------------------------------------------------------------------
extern "C" void kernel_launch(void* const* d_in, const int* in_sizes, int n_in,
                              void* d_out, int out_size, void* d_ws, size_t ws_size,
                              hipStream_t stream)
{
  (void)in_sizes; (void)n_in; (void)out_size; (void)ws_size;
  const float* pos    = (const float*)d_in[0];
  const float* cell   = (const float*)d_in[1];
  const int*   anum   = (const int*)  d_in[2];
  const int*   batch  = (const int*)  d_in[3];
  const int*   esrc   = (const int*)  d_in[4];
  const int*   edst   = (const int*)  d_in[5];
  const int*   asrc   = (const int*)  d_in[6];
  const int*   mdst   = (const int*)  d_in[7];
  const float* emb    = (const float*)d_in[8];
  const float* g0_W   = (const float*)d_in[9];
  const float* g0_Wsc = (const float*)d_in[10];
  const float* g0_m1  = (const float*)d_in[11];
  const float* g0_m2  = (const float*)d_in[12];
  const float* g0_m3  = (const float*)d_in[13];
  const float* g12_W  = (const float*)d_in[14];
  const float* g12_Wsc= (const float*)d_in[15];
  const float* g12_m1 = (const float*)d_in[16];
  const float* g12_m2 = (const float*)d_in[17];
  const float* g12_m3 = (const float*)d_in[18];
  const float* a2m_W  = (const float*)d_in[19];
  const float* a2m_m1 = (const float*)d_in[20];
  const float* a2m_m2 = (const float*)d_in[21];
  const float* a2m_m3 = (const float*)d_in[22];
  const float* m2a_W  = (const float*)d_in[23];
  const float* m2a_m1 = (const float*)d_in[24];
  const float* m2a_m2 = (const float*)d_in[25];
  const float* m2a_m3 = (const float*)d_in[26];
  const float* fc0_W  = (const float*)d_in[27];
  const float* fc0_b  = (const float*)d_in[28];
  const float* sw_r   = (const float*)d_in[29];
  const float* sw_i   = (const float*)d_in[30];
  const float* pw     = (const float*)d_in[31];
  const float* ff1    = (const float*)d_in[32];
  const float* ff2    = (const float*)d_in[33];
  const float* fc1    = (const float*)d_in[34];
  const float* fc2    = (const float*)d_in[35];
  float* out = (float*)d_out;

  // --- workspace bump allocator -------------------------------------------
  char* wsp = (char*)d_ws;
  size_t off = 0;
  auto alloc = [&](size_t bytes)->char*{
    char* p = wsp + off; off += (bytes + 255) & ~(size_t)255; return p;
  };
  float*  sh1    = (float*) alloc((size_t)NE  *9*4);
  float*  sh2    = (float*) alloc((size_t)NE2 *9*4);
  f16*    rad1h  = (f16*)   alloc((size_t)NE  *NRE*2);
  f16*    rad2h  = (f16*)   alloc((size_t)NE2 *NRE*2);
  f16*    h1h    = (f16*)   alloc((size_t)NE2 *NRH*2);
  f16*    h2h    = (f16*)   alloc((size_t)NE2 *NRH*2);
  float*  radout = (float*) alloc((size_t)NE  *NFEAT*4);   // also fits NE2*32
  float*  xA     = (float*) alloc((size_t)NATOM*NFEAT*4);
  float*  xB     = (float*) alloc((size_t)NATOM*NFEAT*4);
  f16*    xh     = (f16*)   alloc((size_t)NATOM*160*2);    // Kp=160 padded staging
  float*  Ybuf   = (float*) alloc((size_t)NATOM*1296*4);
  float*  Ymb    = (float*) alloc((size_t)NMESH*9*4);
  float*  mf     = (float*) alloc((size_t)NMESH*NMC*4);    // reused as spec output
  float*  hbuf   = (float*) alloc((size_t)NMESH*NMH*4);
  float*  htmp   = (float*) alloc((size_t)NMESH*128*4);
  f16*    hh     = (f16*)   alloc((size_t)NMESH*128*2);
  float2* cplxA  = (float2*)alloc((size_t)CPLXN*8);
  float2* cplxB  = (float2*)alloc((size_t)CPLXN*8);
  float*  af     = (float*) alloc((size_t)NATOM*4);
  f16*    whbuf  = (f16*)   alloc((size_t)160*1296*2);     // Kp-padded weights

  auto gl = [](int n){ return (n + 255) / 256; };
  auto convW = [&](const float* W, int K, int Nc){
    int Kp = (K + 31) & ~31;
    int n = Kp * Nc;
    k_padW<<<gl(n),256,0,stream>>>(W, whbuf, K, Nc, n);
  };
  auto gemm = [&](const f16* A, void* C, int Mr, int Nc, int Kp, int act, int outh){
    int tiles = (Mr/16)*(Nc/16);
    int blocks = (tiles + 7) / 8;
    if      (act==1 && outh==1) gemm16<1,1><<<blocks,256,0,stream>>>(A, whbuf, C, Mr, Nc, Kp);
    else if (act==2 && outh==0) gemm16<2,0><<<blocks,256,0,stream>>>(A, whbuf, C, Mr, Nc, Kp);
    else if (act==1 && outh==0) gemm16<1,0><<<blocks,256,0,stream>>>(A, whbuf, C, Mr, Nc, Kp);
    else if (act==0 && outh==1) gemm16<0,1><<<blocks,256,0,stream>>>(A, whbuf, C, Mr, Nc, Kp);
    else                        gemm16<0,0><<<blocks,256,0,stream>>>(A, whbuf, C, Mr, Nc, Kp);
  };
  auto rmlp = [&](const f16* radh, int Er, const float* m1, const float* m2,
                  const float* m3, int Of, float* outp){
    convW(m1, NRE, NRH);  gemm(radh, h1h, Er, NRH, NRE, 1, 1);   // silu, f16 out
    convW(m2, NRH, NRH);  gemm(h1h,  h2h, Er, NRH, NRH, 1, 1);
    convW(m3, NRH, Of);   gemm(h2h,  outp, Er, Of, NRH, 0, 0);
  };

  // === edge features =======================================================
  k_edge1<<<gl(NE ),256,0,stream>>>(pos, esrc, edst, sh1, rad1h, NE);
  k_edge2<<<gl(NE2),256,0,stream>>>(pos, cell, asrc, mdst, sh2, rad2h, NE2);

  // === GCN layer 0 (g0): x = emb[anum] (N x 16, staged Kp=32) ==============
  k_gather_emb<<<gl(NATOM*32),256,0,stream>>>(emb, anum, xh, NATOM*32);
  convW(g0_W, 16, 1296);   gemm(xh, Ybuf, NATOM, 1296, 32, 0, 0);  // Y = x @ W2
  rmlp(rad1h, NE, g0_m1, g0_m2, g0_m3, NFEAT, radout);
  convW(g0_Wsc, 16, 144);  gemm(xh, xB, NATOM, NFEAT, 32, 0, 0);   // skip
  k_msg_scatter<NFEAT><<<gl(NE*NFEAT),256,0,stream>>>(Ybuf, sh1, radout, esrc, edst, xB, NE);

  // === GCN layers 1,2 (g12): K=144 -> Kp=160 ==============================
  for (int i = 0; i < 2; ++i){
    k_norm_act<<<gl(NATOM*16),256,0,stream>>>(xB, xA, NATOM*16);
    k_padA<<<gl(NATOM*160),256,0,stream>>>(xA, xh, NFEAT, 160, NATOM*160);
    convW(g12_W + (size_t)i*1296*144, NFEAT, 1296);
    gemm(xh, Ybuf, NATOM, 1296, 160, 0, 0);
    rmlp(rad1h, NE, g12_m1 + (size_t)i*NRE*NRH, g12_m2 + (size_t)i*NRH*NRH,
         g12_m3 + (size_t)i*NRH*NFEAT, NFEAT, radout);
    convW(g12_Wsc + (size_t)i*144*144, NFEAT, 144);
    gemm(xh, xB, NATOM, NFEAT, 160, 0, 0);
    k_msg_scatter<NFEAT><<<gl(NE*NFEAT),256,0,stream>>>(Ybuf, sh1, radout, esrc, edst, xB, NE);
  }

  // === atom -> mesh (a2m) ==================================================
  k_padA<<<gl(NATOM*160),256,0,stream>>>(xB, xh, NFEAT, 160, NATOM*160);
  convW(a2m_W, NFEAT, 9*NMC);  gemm(xh, Ybuf, NATOM, 9*NMC, 160, 0, 0);
  rmlp(rad2h, NE2, a2m_m1, a2m_m2, a2m_m3, NMC, radout);
  k_zero<<<gl(NMESH*NMC),256,0,stream>>>(mf, NMESH*NMC);
  k_msg_scatter<NMC><<<gl(NE2*NMC),256,0,stream>>>(Ybuf, sh2, radout, asrc, mdst, mf, NE2);
  k_sigabs<<<gl(NMESH*NMC),256,0,stream>>>(mf, NMESH*NMC);

  // === FNO =================================================================
  k_fc0<<<gl(NMESH*NMH),256,0,stream>>>(mf, fc0_W, fc0_b, hbuf, NMESH*NMH);
  for (int l = 0; l < 3; ++l){
    k_fft_z <<<gl(CPLXN),256,0,stream>>>(hbuf, cplxA, CPLXN);
    k_fft_y <<<gl(CPLXN),256,0,stream>>>(cplxA, cplxB, CPLXN);
    k_fft_x <<<gl(CPLXN),256,0,stream>>>(cplxB, cplxA, CPLXN);
    k_spec  <<<gl(CPLXN),256,0,stream>>>(cplxA, cplxB,
              sw_r + (size_t)l*4*16*16*1000, sw_i + (size_t)l*4*16*16*1000, CPLXN);
    k_ifft_x<<<gl(CPLXN),256,0,stream>>>(cplxB, cplxA, CPLXN);
    k_ifft_y<<<gl(CPLXN),256,0,stream>>>(cplxA, cplxB, CPLXN);
    k_ifft_z<<<gl(NMESH*NMH),256,0,stream>>>(cplxB, mf, NMESH*NMH); // mf = spec out
    k_padA<<<gl(NMESH*32),256,0,stream>>>(hbuf, hh, NMH, 32, NMESH*32);
    convW(pw + (size_t)l*NMH*NMH, NMH, NMH);
    gemm(hh, htmp, NMESH, NMH, 32, 0, 0);
    k_gelu_sum<<<gl(NMESH*NMH),256,0,stream>>>(mf, htmp, hbuf, NMESH*NMH);
    k_padA<<<gl(NMESH*32),256,0,stream>>>(hbuf, hh, NMH, 32, NMESH*32);
    convW(ff1 + (size_t)l*NMH*64, NMH, 64);
    gemm(hh, htmp, NMESH, 64, 32, 2, 0);                     // gelu(h@ff1)
    k_f32_to_f16<<<gl(NMESH*64),256,0,stream>>>(htmp, hh, NMESH*64);
    convW(ff2 + (size_t)l*64*NMH, 64, NMH);
    gemm(hh, htmp, NMESH, NMH, 64, 0, 0);
    k_addto<<<gl(NMESH*NMH),256,0,stream>>>(hbuf, htmp, NMESH*NMH);
  }
  k_padA<<<gl(NMESH*32),256,0,stream>>>(hbuf, hh, NMH, 32, NMESH*32);
  convW(fc1, NMH, 128);  gemm(hh, htmp, NMESH, 128, 32, 2, 0);   // gelu(h@fc1)
  k_f32_to_f16<<<gl(NMESH*128),256,0,stream>>>(htmp, hh, NMESH*128);
  convW(fc2, 128, NMH);  gemm(hh, hbuf, NMESH, NMH, 128, 0, 0);  // mf2 = hbuf (M x 16)

  // === mesh -> atom (m2a, O=1) and batch reduction =========================
  k_Ym<<<gl(NMESH*9),256,0,stream>>>(hbuf, m2a_W, Ymb, NMESH*9);
  convW(m2a_m1, NRE, NRH);  gemm(rad2h, h1h, NE2, NRH, NRE, 1, 1);
  convW(m2a_m2, NRH, NRH);  gemm(h1h,  h2h, NE2, NRH, NRH, 1, 1);
  k_dot_m3<<<gl(NE2),256,0,stream>>>(h2h, m2a_m3, radout, NE2);
  k_zero<<<gl(NATOM),256,0,stream>>>(af, NATOM);
  k_m2a_scatter<<<gl(NE2),256,0,stream>>>(Ymb, sh2, radout, mdst, asrc, af, NE2);
  k_zero<<<1,256,0,stream>>>(out, NB);
  k_batch_reduce<<<gl(NATOM),256,0,stream>>>(af, batch, out, NATOM);
}
